// BayesianPredictor_27582279975098
// MI455X (gfx1250) — matrix operand order repaired and checked by reading, not compiled, not run on hardware
//
#include <hip/hip_runtime.h>
#include <hip/hip_bf16.h>
#include <math.h>

// Problem constants (from reference)
#define BB 16384        // batch
#define ID 2048         // in_dim
#define CC 101          // classes
#define CP 112          // classes padded to 7 * 16
#define NT 7            // column tiles of 16
#define NP 10           // passes

#define OFF_LP  ((size_t)BB * CC)      // log_prior scalar
#define OFF_AL  (OFF_LP + 2)           // aleatoric [BB]
#define OFF_EP  (OFF_AL + BB)          // epistemic [BB]

#define LS2PI 0.91893853320467274f     // 0.5*log(2*pi)
#define E12   162754.791419f           // e^12 = 1/sigma2^2

typedef __attribute__((ext_vector_type(16))) __bf16 v16bf;
typedef __attribute__((ext_vector_type(8)))  __bf16 v8bf;
typedef __attribute__((ext_vector_type(8)))  float  v8f;

static __device__ __forceinline__ v16bf cat16(v8bf lo, v8bf hi) {
  return __builtin_shufflevector(lo, hi, 0,1,2,3,4,5,6,7,8,9,10,11,12,13,14,15);
}
static __device__ __forceinline__ v8f vzero8() {
  v8f z = {0.f,0.f,0.f,0.f,0.f,0.f,0.f,0.f};
  return z;
}
// Butterfly reductions across the 16 lanes of each half-wave (wave32).
static __device__ __forceinline__ v8f red16_add(v8f v) {
  #pragma unroll
  for (int m = 1; m < 16; m <<= 1) {
    #pragma unroll
    for (int j = 0; j < 8; ++j) v[j] += __shfl_xor(v[j], m, 32);
  }
  return v;
}
static __device__ __forceinline__ v8f red16_max(v8f v) {
  #pragma unroll
  for (int m = 1; m < 16; m <<= 1) {
    #pragma unroll
    for (int j = 0; j < 8; ++j) v[j] = fmaxf(v[j], __shfl_xor(v[j], m, 32));
  }
  return v;
}

// ---------------- init: zero the scalar atomic accumulators --------------
__global__ void bnn_init(float* __restrict__ out) {
  if (threadIdx.x == 0) { out[OFF_LP] = 0.f; out[OFF_LP + 1] = 0.f; }
}

// ---------------- sample weights -> bf16 padded, log-prob reduce ---------
__global__ __launch_bounds__(256) void bnn_prep_w(
    const float* __restrict__ wmu, const float* __restrict__ wrho,
    const float* __restrict__ ew, __bf16* __restrict__ wbd,
    float* __restrict__ out)
{
  const int c = blockIdx.x;        // 0..111 (padded)
  const int n = blockIdx.y;        // 0..9
  const int tid = threadIdx.x;
  __bf16* dst = wbd + ((size_t)n * CP + c) * ID;
  float lp = 0.f, lq = 0.f;
  if (c < CC) {
    const float* mu  = wmu  + (size_t)c * ID;
    const float* rho = wrho + (size_t)c * ID;
    const float* ev  = ew + ((size_t)n * CC + c) * ID;
    for (int i = tid; i < ID; i += 256) {
      float sg = log1pf(__expf(rho[i]));
      float e  = ev[i];
      float v  = mu[i] + sg * e;
      dst[i] = (__bf16)v;
      float v2  = v * v;
      float lp1 = -LS2PI - 0.5f * v2;                  // N(0, 1)
      float lp2 = -LS2PI + 6.0f - v2 * (0.5f * E12);   // N(0, e^-6)
      float mm  = fmaxf(lp1, lp2);
      lp += mm + logf(0.5f * (__expf(lp1 - mm) + __expf(lp2 - mm)));
      lq += -LS2PI - logf(sg) - 0.5f * e * e;          // (v-mu)/sg = eps
    }
  } else {
    for (int i = tid; i < ID; i += 256) dst[i] = (__bf16)0.f;
  }
  #pragma unroll
  for (int m = 1; m < 32; m <<= 1) { lp += __shfl_xor(lp, m, 32); lq += __shfl_xor(lq, m, 32); }
  __shared__ float shp[8], shq[8];
  const int lane = tid & 31, wid = tid >> 5;
  if (lane == 0) { shp[wid] = lp; shq[wid] = lq; }
  __syncthreads();
  if (tid == 0) {
    float a = 0.f, b = 0.f;
    for (int w2 = 0; w2 < 8; ++w2) { a += shp[w2]; b += shq[w2]; }
    atomicAdd(&out[OFF_LP],     a * (1.0f / NP));
    atomicAdd(&out[OFF_LP + 1], b * (1.0f / NP));
  }
}

// ---------------- sample biases, log-prob reduce -------------------------
__global__ __launch_bounds__(128) void bnn_prep_b(
    const float* __restrict__ bmu, const float* __restrict__ brho,
    const float* __restrict__ eb, float* __restrict__ bsd,
    float* __restrict__ out)
{
  const int tid = threadIdx.x;
  float lp = 0.f, lq = 0.f;
  for (int idx = tid; idx < NP * CC; idx += 128) {
    const int n = idx / CC;
    const int c = idx - n * CC;
    float sg = log1pf(__expf(brho[c]));
    float e  = eb[idx];
    float v  = bmu[c] + sg * e;
    bsd[idx] = v;
    float v2  = v * v;
    float lp1 = -LS2PI - 0.5f * v2;
    float lp2 = -LS2PI + 6.0f - v2 * (0.5f * E12);
    float mm  = fmaxf(lp1, lp2);
    lp += mm + logf(0.5f * (__expf(lp1 - mm) + __expf(lp2 - mm)));
    lq += -LS2PI - logf(sg) - 0.5f * e * e;
  }
  #pragma unroll
  for (int m = 1; m < 32; m <<= 1) { lp += __shfl_xor(lp, m, 32); lq += __shfl_xor(lq, m, 32); }
  __shared__ float shp[4], shq[4];
  const int lane = tid & 31, wid = tid >> 5;
  if (lane == 0) { shp[wid] = lp; shq[wid] = lq; }
  __syncthreads();
  if (tid == 0) {
    float a = 0.f, b = 0.f;
    for (int w2 = 0; w2 < 4; ++w2) { a += shp[w2]; b += shq[w2]; }
    atomicAdd(&out[OFF_LP],     a * (1.0f / NP));
    atomicAdd(&out[OFF_LP + 1], b * (1.0f / NP));
  }
}

// ---------------- convert x f32 -> bf16 ----------------------------------
__global__ __launch_bounds__(256) void bnn_conv_x(
    const float* __restrict__ x, __bf16* __restrict__ xbd)
{
  const size_t i = ((size_t)blockIdx.x * 256 + threadIdx.x) * 8;
  const float4 a = *(const float4*)(x + i);
  const float4 b = *(const float4*)(x + i + 4);
  v8bf o;
  o[0] = (__bf16)a.x; o[1] = (__bf16)a.y; o[2] = (__bf16)a.z; o[3] = (__bf16)a.w;
  o[4] = (__bf16)b.x; o[5] = (__bf16)b.y; o[6] = (__bf16)b.z; o[7] = (__bf16)b.w;
  *(v8bf*)(xbd + i) = o;
}

// ---------------- fused WMMA GEMM + softmax statistics -------------------
// Block = 128 threads = 4 waves; each wave owns 16 rows of x and the full
// padded class dimension (7 WMMA tiles of 16). Pass loop inside the kernel:
// per pass the wave computes logits with v_wmma_f32_16x16x32_bf16, then does
// softmax + stats in-register via 16-lane butterflies (D layout:
// col = lane&15, row = j + 8*(lane>>4)).
__global__ __launch_bounds__(128) void bnn_gemm_fused(
    const __bf16* __restrict__ xb, const __bf16* __restrict__ wb,
    const float* __restrict__ bs, float* __restrict__ out)
{
  const int lane = threadIdx.x & 31;
  const int wave = threadIdx.x >> 5;
  const int half = lane >> 4;
  const int l15  = lane & 15;
  const int row0 = (blockIdx.x * 4 + wave) * 16;

  const __bf16* xrow = xb + (size_t)(row0 + l15) * ID;
  const int aoff0 = half * 16;       // A elements 0..7  = K aoff0..+7
  const int aoff1 = half * 16 + 8;   // A elements 8..15 = K aoff1..+7
  const int boff  = half * 16;       // B: 16 contiguous K starting here

  v8f sumL[NT], s1[NT], s2[NT], al = vzero8();
  #pragma unroll
  for (int t = 0; t < NT; ++t) { sumL[t] = vzero8(); s1[t] = vzero8(); s2[t] = vzero8(); }

  for (int n = 0; n < NP; ++n) {
    const __bf16* wn = wb + (size_t)n * CP * ID;
    const __bf16* wrt[NT];
    #pragma unroll
    for (int t = 0; t < NT; ++t) wrt[t] = wn + (size_t)(t * 16 + l15) * ID + boff;

    v8f acc[NT];
    #pragma unroll
    for (int t = 0; t < NT; ++t) acc[t] = vzero8();

    #pragma unroll 4
    for (int k = 0; k < ID; k += 32) {
      v16bf A = cat16(*(const v8bf*)(xrow + k + aoff0),
                      *(const v8bf*)(xrow + k + aoff1));
      #pragma unroll
      for (int t = 0; t < NT; ++t) {
        v16bf Bt = cat16(*(const v8bf*)(wrt[t] + k),
                         *(const v8bf*)(wrt[t] + k + 8));
        acc[t] = __builtin_amdgcn_wmma_f32_16x16x32_bf16(
            false, A, false, Bt, (short)0, acc[t], false, false);
      }
    }

    // bias + invalid-column mask
    v8f logit[NT];
    #pragma unroll
    for (int t = 0; t < NT; ++t) {
      const int c = t * 16 + l15;
      if (c < CC) {
        logit[t] = acc[t] + bs[n * CC + c];
      } else {
        #pragma unroll
        for (int j = 0; j < 8; ++j) logit[t][j] = -3.0e30f;
      }
    }

    // row-wise softmax over 112 (padded) columns
    v8f mx = logit[0];
    #pragma unroll
    for (int t = 1; t < NT; ++t) {
      #pragma unroll
      for (int j = 0; j < 8; ++j) mx[j] = fmaxf(mx[j], logit[t][j]);
    }
    mx = red16_max(mx);

    v8f p[NT], ps = vzero8();
    #pragma unroll
    for (int t = 0; t < NT; ++t) {
      #pragma unroll
      for (int j = 0; j < 8; ++j) p[t][j] = __expf(logit[t][j] - mx[j]);
      ps += p[t];
    }
    ps = red16_add(ps);
    v8f inv;
    #pragma unroll
    for (int j = 0; j < 8; ++j) inv[j] = 1.0f / ps[j];

    v8f q1 = vzero8(), q2 = vzero8();
    #pragma unroll
    for (int t = 0; t < NT; ++t) {
      #pragma unroll
      for (int j = 0; j < 8; ++j) {
        float pn = p[t][j] * inv[j];
        q1[j] += pn; q2[j] += pn * pn;
        s1[t][j] += pn; s2[t][j] += pn * pn;
      }
      sumL[t] += logit[t];
    }
    q1 = red16_add(q1); q2 = red16_add(q2);
    al += q1 - q2;                           // per-row aleatoric contribution
  }

  // ---- epilogue ----
  const float invN = 1.0f / NP;
  #pragma unroll
  for (int t = 0; t < NT; ++t) {
    const int c = t * 16 + l15;
    if (c < CC) {
      #pragma unroll
      for (int j = 0; j < 8; ++j)
        out[(size_t)(row0 + j + 8 * half) * CC + c] = sumL[t][j] * invN;
    }
  }
  v8f ep = vzero8();
  #pragma unroll
  for (int t = 0; t < NT; ++t) {
    #pragma unroll
    for (int j = 0; j < 8; ++j) {
      float a = s1[t][j] * invN;
      ep[j] += s2[t][j] * invN - a * a;   // sum_c (S2/n - pbar^2)
    }
  }
  ep = red16_add(ep);
  if (l15 == 0) {
    #pragma unroll
    for (int j = 0; j < 8; ++j) {
      const int r = row0 + j + 8 * half;
      out[OFF_AL + r] = al[j] * invN;
      out[OFF_EP + r] = ep[j];
    }
  }
}

extern "C" void kernel_launch(void* const* d_in, const int* in_sizes, int n_in,
                              void* d_out, int out_size, void* d_ws, size_t ws_size,
                              hipStream_t stream) {
  (void)in_sizes; (void)n_in; (void)out_size; (void)ws_size;
  const float* x    = (const float*)d_in[0];
  const float* wmu  = (const float*)d_in[1];
  const float* wrho = (const float*)d_in[2];
  const float* bmu  = (const float*)d_in[3];
  const float* brho = (const float*)d_in[4];
  const float* ew   = (const float*)d_in[5];
  const float* eb   = (const float*)d_in[6];
  float* out = (float*)d_out;

  char* ws = (char*)d_ws;
  __bf16* xbd = (__bf16*)ws;                                   // 64 MiB
  __bf16* wbd = (__bf16*)(ws + (size_t)BB * ID * 2);           // 4.4 MiB
  float*  bsd = (float*)(ws + (size_t)BB * ID * 2
                            + (size_t)NP * CP * ID * 2);       // 4 KiB

  bnn_init<<<1, 32, 0, stream>>>(out);
  bnn_prep_w<<<dim3(CP, NP), 256, 0, stream>>>(wmu, wrho, ew, wbd, out);
  bnn_prep_b<<<1, 128, 0, stream>>>(bmu, brho, eb, bsd, out);
  bnn_conv_x<<<(BB * ID) / (256 * 8), 256, 0, stream>>>(x, xbd);
  bnn_gemm_fused<<<BB / 64, 128, 0, stream>>>(xbd, wbd, bsd, out);
}
